// NNCRF_5308579578356
// MI455X (gfx1250) — matrix operand order repaired and checked by reading, not compiled, not run on hardware
//
#include <hip/hip_runtime.h>

typedef __attribute__((ext_vector_type(16))) _Float16 v16h;
typedef __attribute__((ext_vector_type(8)))  float    v8f;
typedef __attribute__((ext_vector_type(2)))  __fp16   fp16x2;

#define BATCH  2048
#define S_LEN  1024
#define L_TAGS 32

#define LOG2E 1.4426950408889634f
#define LN2   0.6931471805599453f

union HPack { unsigned u; _Float16 h[2]; };

// ---- raw hardware transcendentals (base-2, no denorm-guard expansion) -----
__device__ __forceinline__ float fast_exp2(float x) {
#if __has_builtin(__builtin_amdgcn_exp2f)
    return __builtin_amdgcn_exp2f(x);
#else
    return exp2f(x);
#endif
}
__device__ __forceinline__ float fast_log2(float x) {
#if __has_builtin(__builtin_amdgcn_logf)
    return __builtin_amdgcn_logf(x);
#else
    return log2f(x);
#endif
}

// ---- xor-16 lane exchange: v_permlanex16_b32 (VALU) instead of ds_bpermute
__device__ __forceinline__ unsigned xor16_u(unsigned v) {
#if __has_builtin(__builtin_amdgcn_permlanex16)
    return (unsigned)__builtin_amdgcn_permlanex16((int)v, (int)v,
                                                  0x76543210, 0xFEDCBA98,
                                                  false, false);
#else
    return (unsigned)__shfl_xor((int)v, 16, 32);
#endif
}
__device__ __forceinline__ float xor16_f(float v) {
    union { float f; unsigned u; } a; a.f = v;
    a.u = xor16_u(a.u);
    return a.f;
}

// ---- packed f32x2 -> f16x2 convert (v_cvt_pk_rtz_f16_f32) -----------------
__device__ __forceinline__ unsigned pk_f16(float a, float b) {
#if __has_builtin(__builtin_amdgcn_cvt_pkrtz)
    union { fp16x2 h; unsigned u; } c;
    c.h = __builtin_amdgcn_cvt_pkrtz(a, b);
    return c.u;
#else
    HPack p; p.h[0] = (_Float16)a; p.h[1] = (_Float16)b; return p.u;
#endif
}

__device__ __forceinline__ void load8(const float* __restrict__ p, float* d) {
    float4 v0 = *(const float4*)p;
    float4 v1 = *(const float4*)(p + 4);
    d[0]=v0.x; d[1]=v0.y; d[2]=v0.z; d[3]=v0.w;
    d[4]=v1.x; d[5]=v1.y; d[6]=v1.z; d[7]=v1.w;
}

// ---------------------------------------------------------------------------
// Unlabeled (log-partition) kernel, log2-domain state beta = alpha * log2(e).
// One wave32 per 16-batch tile.  beta^T[j,b]: lane = column b (lanes L and
// L^16 share a column), VGPR r = row j = r + 8*(lane>=16) (+16 for "hi"),
// matching the WMMA f32 16x16 C/D layout.  Per step:
//   C = A x B,  A = 2^(T^T * log2e) (constant f16 16x32 frags in regs),
//   B = 2^(beta - rowmax) (f16 32x16, built with one permlanex16 per pair)
//   beta' = rowmax + log2(C) + emit * log2e
// ---------------------------------------------------------------------------
__global__ void __launch_bounds__(32)
crf_unlabeled_kernel(const float* __restrict__ scores,
                     const float* __restrict__ trans,
                     const int*   __restrict__ lens,
                     float*       __restrict__ out)
{
    const int lane = threadIdx.x & 31;
    const int g    = lane >> 4;        // half-wave: selects which 8 rows we own
    const int col  = lane & 15;        // batch column within tile
    const int b    = blockIdx.x * 16 + col;
    const int myLen = lens[b];
    const int startIdx = L_TAGS - 1, endIdx = L_TAGS - 2;

    // --- constant A fragments: A_tile[m, k] = 2^(T[k, jglobal(m)] * log2e) --
    // f16 16x32 A, this code's k order (as B): lane holds row m = lane%16,
    // half p -> K = p + 16g.
    v16h A_lo, A_hi;
    #pragma unroll
    for (int p = 0; p < 16; ++p) {
        int k = p + 16 * g;
        A_lo[p] = (_Float16)fast_exp2(trans[k * L_TAGS + col]      * LOG2E);
        A_hi[p] = (_Float16)fast_exp2(trans[k * L_TAGS + 16 + col] * LOG2E);
    }

    // per-lane constants (already scaled to log2 domain)
    float tEnd_lo[8], tEnd_hi[8], tStart_lo[8], tStart_hi[8];
    #pragma unroll
    for (int r = 0; r < 8; ++r) {
        int jlo = r + 8 * g, jhi = 16 + r + 8 * g;
        tEnd_lo[r]   = trans[jlo * L_TAGS + endIdx] * LOG2E;
        tEnd_hi[r]   = trans[jhi * L_TAGS + endIdx] * LOG2E;
        tStart_lo[r] = trans[startIdx * L_TAGS + jlo] * LOG2E;
        tStart_hi[r] = trans[startIdx * L_TAGS + jhi] * LOG2E;
    }

    const float* rowPtr = scores + (size_t)b * S_LEN * L_TAGS;

    // --- beta0 = (T[start,:] + emit[:,0,:]) * log2e -------------------------
    float a_lo[8], a_hi[8];
    {
        float elo[8], ehi[8];
        load8(rowPtr + 8 * g, elo);
        load8(rowPtr + 16 + 8 * g, ehi);
        #pragma unroll
        for (int r = 0; r < 8; ++r) {
            a_lo[r] = fmaf(elo[r], LOG2E, tStart_lo[r]);
            a_hi[r] = fmaf(ehi[r], LOG2E, tStart_hi[r]);
        }
    }

    // capture of last_alpha contribution (lanes L and L^16 share b => uniform
    // within each xor-16 pair, so the paired exchanges are safe under the if)
    auto do_capture = [&](int t) {
        if (t == myLen - 1) {
            float mx = -1e30f;
            float v_lo[8], v_hi[8];
            #pragma unroll
            for (int r = 0; r < 8; ++r) {
                v_lo[r] = a_lo[r] + tEnd_lo[r];
                v_hi[r] = a_hi[r] + tEnd_hi[r];
                mx = fmaxf(mx, fmaxf(v_lo[r], v_hi[r]));
            }
            mx = fmaxf(mx, xor16_f(mx));
            float s = 0.f;
            #pragma unroll
            for (int r = 0; r < 8; ++r)
                s += fast_exp2(v_lo[r] - mx) + fast_exp2(v_hi[r] - mx);
            s += xor16_f(s);
            if (g == 0) atomicAdd(out + 0, (mx + fast_log2(s)) * LN2);
        }
    };

    do_capture(0);

    for (int t = 1; t < S_LEN; ++t) {
        const float* tp = rowPtr + (size_t)t * L_TAGS;
        if (t + 6 < S_LEN)
            __builtin_prefetch(rowPtr + (size_t)(t + 6) * L_TAGS, 0, 1);

        float elo[8], ehi[8];
        load8(tp + 8 * g, elo);
        load8(tp + 16 + 8 * g, ehi);

        // row-max over all 32 j of this column
        float m = a_lo[0];
        #pragma unroll
        for (int r = 0; r < 8; ++r) m = fmaxf(m, fmaxf(a_lo[r], a_hi[r]));
        float mm = fmaxf(m, xor16_f(m));

        // 2^(beta - mm) -> packed f16 pairs; exchange with partner lane
        unsigned ulo[4], uhi[4], plo[4], phi[4];
        #pragma unroll
        for (int r = 0; r < 4; ++r) {
            ulo[r] = pk_f16(fast_exp2(a_lo[2*r]     - mm),
                            fast_exp2(a_lo[2*r + 1] - mm));
            uhi[r] = pk_f16(fast_exp2(a_hi[2*r]     - mm),
                            fast_exp2(a_hi[2*r + 1] - mm));
        }
        #pragma unroll
        for (int r = 0; r < 4; ++r) {
            plo[r] = xor16_u(ulo[r]);
            phi[r] = xor16_u(uhi[r]);
        }
        // B f16 32x16, this code's k order (as A): lanes 0-15 hold K=0..15 (2/VGPR), lanes 16-31 K=16..31
        union { v16h h; unsigned u[8]; } Bm;
        #pragma unroll
        for (int r = 0; r < 4; ++r) {
            Bm.u[r]     = g ? phi[r]   : ulo[r];  // K = 0..7   / 16..23
            Bm.u[r + 4] = g ? uhi[r]   : plo[r];  // K = 8..15  / 24..31
        }

        v8f c0 = {}; v8f c1 = {};
        c0 = __builtin_amdgcn_wmma_f32_16x16x32_f16(false, A_lo, false, Bm.h,
                                                    (short)0, c0, false, false);
        c1 = __builtin_amdgcn_wmma_f32_16x16x32_f16(false, A_hi, false, Bm.h,
                                                    (short)0, c1, false, false);

        #pragma unroll
        for (int r = 0; r < 8; ++r) {
            a_lo[r] = fmaf(elo[r], LOG2E, mm + fast_log2(c0[r]));
            a_hi[r] = fmaf(ehi[r], LOG2E, mm + fast_log2(c1[r]));
        }

        do_capture(t);
    }
}

// ---------------------------------------------------------------------------
// Labeled path score: one block per batch row, threads stride over time.
// ---------------------------------------------------------------------------
__global__ void __launch_bounds__(256)
crf_labeled_kernel(const float* __restrict__ scores,
                   const float* __restrict__ trans,
                   const int*   __restrict__ lens,
                   const int*   __restrict__ tags,
                   float*       __restrict__ out)
{
    __shared__ float sT[L_TAGS * L_TAGS];
    __shared__ float red[256];
    const int b = blockIdx.x;
    for (int i = threadIdx.x; i < L_TAGS * L_TAGS; i += 256) sT[i] = trans[i];
    __syncthreads();

    const int len = lens[b];
    const int* tg = tags + (size_t)b * S_LEN;
    const float* sc = scores + (size_t)b * S_LEN * L_TAGS;
    const int startIdx = L_TAGS - 1, endIdx = L_TAGS - 2;

    float acc = 0.f;
    for (int t = threadIdx.x; t < len; t += 256) {
        int cur = tg[t];
        float emit = sc[(size_t)t * L_TAGS + cur];
        float tr = (t == 0) ? sT[startIdx * L_TAGS + cur]
                            : sT[tg[t - 1] * L_TAGS + cur];
        acc += tr + emit;
        if (t == len - 1) acc += sT[cur * L_TAGS + endIdx];
    }
    red[threadIdx.x] = acc;
    __syncthreads();
    for (int s = 128; s > 0; s >>= 1) {
        if (threadIdx.x < s) red[threadIdx.x] += red[threadIdx.x + s];
        __syncthreads();
    }
    if (threadIdx.x == 0) atomicAdd(out + 1, red[0]);
}

__global__ void zero_out_kernel(float* __restrict__ out) {
    if (threadIdx.x < 2) out[threadIdx.x] = 0.f;
}

extern "C" void kernel_launch(void* const* d_in, const int* in_sizes, int n_in,
                              void* d_out, int out_size, void* d_ws, size_t ws_size,
                              hipStream_t stream) {
    const float* scores = (const float*)d_in[0];   // (B, S, L) f32
    const float* trans  = (const float*)d_in[1];   // (L, L)    f32
    const int*   lens   = (const int*)d_in[2];     // (B,)      i32
    const int*   tags   = (const int*)d_in[3];     // (B, S)    i32
    // d_in[4] = mask (derivable from lens; unused)
    float* out = (float*)d_out;                    // [unlabeled, labeled]

    hipLaunchKernelGGL(zero_out_kernel, dim3(1), dim3(32), 0, stream, out);
    hipLaunchKernelGGL(crf_unlabeled_kernel, dim3(BATCH / 16), dim3(32), 0, stream,
                       scores, trans, lens, out);
    hipLaunchKernelGGL(crf_labeled_kernel, dim3(BATCH), dim3(256), 0, stream,
                       scores, trans, lens, tags, out);
}
